// QuantizedLoRALinear_74844100100831
// MI455X (gfx1250) — matrix operand order repaired and checked by reading, not compiled, not run on hardware
//
#include <hip/hip_runtime.h>
#include <stdint.h>

// Problem constants (from reference)
#define IN_F   4096
#define OUT_F  11008
#define LORA_R 16
#define NTOK   64
#define SCALING 2.0f

// Workspace layout:
//   [0, 512KB)      : x packed as bf16 in WMMA A-fragment order (131072 dwords)
//   [512KB, 516KB)  : t = (x @ lora_A^T) * SCALING, [64][16] f32
#define XFRAG_DWORDS (NTOK * IN_F / 2)          // 131072
#define T_OFFSET_BYTES (XFRAG_DWORDS * 4)       // 524288

typedef __attribute__((ext_vector_type(16))) __bf16 bf16x16;
typedef __attribute__((ext_vector_type(8)))  float  floatx8;

// Round-to-nearest-even f32 -> bf16 (bit pattern), no reliance on __bf16 conversions.
static __device__ __forceinline__ uint32_t f32_to_bf16_bits(float f) {
  union { float f; uint32_t u; } c; c.f = f;
  uint32_t u = c.u + 0x7FFFu + ((c.u >> 16) & 1u);
  return u >> 16;
}
static __device__ __forceinline__ uint32_t pack_bf16x2(float lo, float hi) {
  return f32_to_bf16_bits(lo) | (f32_to_bf16_bits(hi) << 16);
}

// -----------------------------------------------------------------------------
// Kernel 1: pack x[64][4096] f32 into bf16 A-fragments in WMMA lane order.
// Index: xfrag[ ((kstep*4 + mtile)*32 + lane)*8 + v ]  (dword holds K pair k,k+1)
// A 16x32 bf16 layout used here (contiguous k, same as B): lane group g = lane>>4, m = mtile*16 + (lane&15)
//   v in 0..7 : k = kstep*32 + g*16 + 2v
// -----------------------------------------------------------------------------
__global__ void qlora_pack_x(const float* __restrict__ x, uint32_t* __restrict__ xfrag) {
  int idx = blockIdx.x * blockDim.x + threadIdx.x;      // 0 .. 131071
  int v     = idx & 7;
  int lane  = (idx >> 3) & 31;
  int mtile = (idx >> 8) & 3;
  int kstep = idx >> 10;
  int m   = mtile * 16 + (lane & 15);
  int grp = lane >> 4;
  int k   = kstep * 32 + grp * 16 + 2 * v;
  float2 f = *(const float2*)(x + (size_t)m * IN_F + k);  // k always even -> 8B aligned
  xfrag[idx] = pack_bf16x2(f.x, f.y);
}

// -----------------------------------------------------------------------------
// Kernel 2: t[64][16] = (x @ lora_A^T) * SCALING  (f32, tiny)
// -----------------------------------------------------------------------------
__global__ void qlora_lora_t(const float* __restrict__ x, const float* __restrict__ loraA,
                             float* __restrict__ t) {
  int tid = blockIdx.x * blockDim.x + threadIdx.x;      // 0 .. 1023
  int m = tid >> 4, r = tid & 15;
  const float4* xr = (const float4*)(x + (size_t)m * IN_F);
  const float4* ar = (const float4*)(loraA + (size_t)r * IN_F);
  float acc = 0.0f;
  for (int i = 0; i < IN_F / 4; ++i) {
    float4 a = xr[i], b = ar[i];
    acc += a.x * b.x + a.y * b.y + a.z * b.z + a.w * b.w;
  }
  t[m * LORA_R + r] = acc * SCALING;
}

// -----------------------------------------------------------------------------
// Kernel 3: main GEMM. 64 threads (2 waves); each wave owns a unique 16-wide N
// strip and all 4 M-tiles (M=64). B fragment dequantized once, reused 4x.
// D layout (f32 16x16): VGPR i -> lanes 0-15: M=i, lanes 16-31: M=8+i; N=lane&15.
// -----------------------------------------------------------------------------
__global__ __launch_bounds__(64) void qlora_gemm(
    const uint32_t* __restrict__ xfrag, const int* __restrict__ wq,
    const float* __restrict__ wscale, const float* __restrict__ loraB,
    const float* __restrict__ t, float* __restrict__ out) {

  __shared__ float t_lds[NTOK * LORA_R];                // 4 KB
  int tid = threadIdx.x;
  for (int i = tid; i < NTOK * LORA_R; i += 64) t_lds[i] = t[i];
  __syncthreads();

  int wave = tid >> 5;
  int lane = tid & 31;
  int nBase = (blockIdx.x * 2 + wave) * 16;             // 344 blocks * 32 = 11008
  int ncol  = lane & 15;
  int n     = nBase + ncol;
  int grp   = lane >> 4;

  float scale = wscale[n];

  // B source: weight row n, K offset grp*16 within each 32-wide K step.
  // Each lane covers a contiguous 64B of its row per K-step -> 4x b128 loads.
  const int* wrow = wq + (size_t)n * IN_F + grp * 16;
  // A source: per-lane contiguous 8-dword fragments
  const uint32_t* abase = xfrag + (size_t)lane * 8;

  union CU { floatx8 v; float f[8]; } acc[4];
  acc[0].v = (floatx8)0.0f; acc[1].v = (floatx8)0.0f;
  acc[2].v = (floatx8)0.0f; acc[3].v = (floatx8)0.0f;

  for (int kstep = 0; kstep < IN_F / 32; ++kstep) {
    const int* wp = wrow + kstep * 32;
    // stream the weight ~1KB ahead (speculative global_prefetch_b8; OOB is dropped)
    __builtin_prefetch((const void*)(wp + 8 * 32), 0, 0);

    // ---- B fragment: 16(N) x 32(K) bf16, dequant int8-range -> exact bf16 ----
    // VGPR v holds K pair (grp*16 + 2v, +1); int4 load gives 4 consecutive K.
    union BU { bf16x16 v; uint32_t u[8]; } bf;
#pragma unroll
    for (int v4 = 0; v4 < 4; ++v4) {
      int4 q = *(const int4*)(wp + 4 * v4);             // 16B aligned
      bf.u[2 * v4]     = pack_bf16x2((float)q.x, (float)q.y);
      bf.u[2 * v4 + 1] = pack_bf16x2((float)q.z, (float)q.w);
    }

    // ---- A fragments: 4 M-tiles, contiguous 32B per lane each ----
    const uint32_t* ab = abase + (size_t)kstep * (4 * 32 * 8);
    union AU { bf16x16 v; uint4 q[2]; } a[4];
#pragma unroll
    for (int mt = 0; mt < 4; ++mt) {
      const uint4* ap = (const uint4*)(ab + mt * (32 * 8));
      a[mt].q[0] = ap[0];
      a[mt].q[1] = ap[1];
    }

    // ---- 4 WMMAs: D = A x B + C, f32 accumulate ----
#pragma unroll
    for (int mt = 0; mt < 4; ++mt) {
      acc[mt].v = __builtin_amdgcn_wmma_f32_16x16x32_bf16(
          false, a[mt].v, false, bf.v, (short)0, acc[mt].v, false, false);
    }
  }

  // ---- Epilogue: out = acc*scale + dot16(t[m], loraB[n]) ----
  const float4* bRow = (const float4*)(loraB + (size_t)n * LORA_R);
  float4 lb0 = bRow[0], lb1 = bRow[1], lb2 = bRow[2], lb3 = bRow[3];

#pragma unroll
  for (int mt = 0; mt < 4; ++mt) {
#pragma unroll
    for (int i = 0; i < 8; ++i) {
      int m = mt * 16 + i + grp * 8;                    // D-matrix VGPR layout
      const float4* tr = (const float4*)(&t_lds[m * LORA_R]);
      float4 t0 = tr[0], t1 = tr[1], t2 = tr[2], t3 = tr[3];
      float lora = t0.x * lb0.x + t0.y * lb0.y + t0.z * lb0.z + t0.w * lb0.w
                 + t1.x * lb1.x + t1.y * lb1.y + t1.z * lb1.z + t1.w * lb1.w
                 + t2.x * lb2.x + t2.y * lb2.y + t2.z * lb2.z + t2.w * lb2.w
                 + t3.x * lb3.x + t3.y * lb3.y + t3.z * lb3.z + t3.w * lb3.w;
      out[(size_t)m * OUT_F + n] = acc[mt].f[i] * scale + lora;
    }
  }
}

// -----------------------------------------------------------------------------
extern "C" void kernel_launch(void* const* d_in, const int* in_sizes, int n_in,
                              void* d_out, int out_size, void* d_ws, size_t ws_size,
                              hipStream_t stream) {
  const float* x      = (const float*)d_in[0];
  const int*   wq     = (const int*)d_in[1];
  const float* wscale = (const float*)d_in[2];
  const float* loraA  = (const float*)d_in[3];
  const float* loraB  = (const float*)d_in[4];
  float* out = (float*)d_out;

  uint32_t* xfrag = (uint32_t*)d_ws;                        // 512 KB
  float*    t     = (float*)((char*)d_ws + T_OFFSET_BYTES); // 4 KB

  // 1) pack x into bf16 WMMA A-fragment layout
  qlora_pack_x<<<XFRAG_DWORDS / 256, 256, 0, stream>>>(x, xfrag);
  // 2) LoRA low-rank intermediate t = (x @ A^T) * SCALING
  qlora_lora_t<<<(NTOK * LORA_R) / 256, 256, 0, stream>>>(x, loraA, t);
  // 3) fused dequant GEMM + LoRA epilogue
  qlora_gemm<<<OUT_F / 32, 64, 0, stream>>>(xfrag, wq, wscale, loraB, t, out);
}
